// SwinTransformerBlock_90520730731135
// MI455X (gfx1250) — matrix operand order repaired and checked by reading, not compiled, not run on hardware
//
#include <hip/hip_runtime.h>

typedef __attribute__((ext_vector_type(16))) __bf16 v16bf;
typedef __attribute__((ext_vector_type(8)))  __bf16 v8bf;
typedef __attribute__((ext_vector_type(8)))  float  v8f;

#define HEADS 6
#define C     192
#define C3    576
#define NTOK  64   // tokens per 8x8 window
#define NWIN  2048 // 8 * 16 * 16

// ---------- helpers ----------

__device__ __forceinline__ __bf16 f2bf(float f) {
  unsigned u = __builtin_bit_cast(unsigned, f);
  u += 0x7FFFu + ((u >> 16) & 1u);           // round-to-nearest-even
  unsigned short h = (unsigned short)(u >> 16);
  return __builtin_bit_cast(__bf16, h);
}

// A fragment: 16x32 (MxK), row-major LDS source, ld = row stride (elements).
// Lane L: m = L%16; hs = L/16 selects contiguous K-half (k = hs*16 + s, same order as B).
// All call sites guarantee 16-byte alignment of both halves.
__device__ __forceinline__ v16bf load_A(const __bf16* base, int row0, int k0, int ld, int lane) {
  int m = lane & 15, hs = lane >> 4;
  const __bf16* r = base + (size_t)(row0 + m) * ld + k0 + hs * 16;
  v8bf lo = *(const v8bf*)(r);        // K = hs*16 .. hs*16+7
  v8bf hi = *(const v8bf*)(r + 8);    // K = hs*16+8 .. hs*16+15
  return __builtin_shufflevector(lo, hi, 0, 1, 2, 3, 4, 5, 6, 7,
                                         8, 9, 10, 11, 12, 13, 14, 15);
}

// B fragment from pre-swizzled weights: tile = 32 lanes x 16 contiguous bf16.
// One coalesced 32-byte load per lane.
__device__ __forceinline__ v16bf load_Bsw(const __bf16* wsw, int tile, int lane) {
  return ((const v16bf*)wsw)[(size_t)tile * 32 + lane];
}

// B fragment: 32x16 (KxN), row-major LDS source with row stride ld. Scalar (strided).
__device__ __forceinline__ v16bf load_B_lds(const __bf16* base, int k0, int col0, int ld, int lane) {
  int n = lane & 15, hs = lane >> 4;
  v16bf b;
#pragma unroll
  for (int i = 0; i < 16; ++i) b[i] = base[(size_t)(k0 + hs * 16 + i) * ld + col0 + n];
  return b;
}

// B fragment where logical B[k][n] = base[n*ld + k] (k contiguous, 32-byte aligned).
__device__ __forceinline__ v16bf load_Bt(const __bf16* base, int col0, int ld, int lane) {
  int n = lane & 15, hs = lane >> 4;
  const __bf16* r = base + (size_t)(col0 + n) * ld + hs * 16;
  return *(const v16bf*)r;
}

__device__ __forceinline__ v8f wmma_bf16(v16bf a, v16bf b, v8f c) {
  return __builtin_amdgcn_wmma_f32_16x16x32_bf16(false, a, false, b, (short)0, c, false, false);
}

// C/D layout: VGPR r -> M = r + 8*(lane/16), N = lane%16
__device__ __forceinline__ void store_C_bf(__bf16* base, int row0, int col0, int ld, int lane, v8f c) {
  int n = lane & 15, hs = lane >> 4;
#pragma unroll
  for (int r = 0; r < 8; ++r)
    base[(size_t)(row0 + r + 8 * hs) * ld + col0 + n] = f2bf(c[r]);
}

__device__ __forceinline__ int rcls(int t) {            // Swin region class, H=128, ws=8, shift=4
  return t < 120 ? 0 : (t < 124 ? 1 : 2);
}

// ---------- kernel 1: fp32 -> bf16 weight convert + WMMA-B-fragment swizzle ----------
// dst layout: tile t = kk*ntiles + nt; within tile: lane (0..31) x 16 contiguous bf16,
// element (lane, i) = W[kk*32 + (lane>>4)*16 + i][nt*16 + (lane&15)].

__global__ void swin_swizzle_w(const float* __restrict__ src, __bf16* __restrict__ dst,
                               int ktiles, int ntiles, int ldn) {
  int idx = blockIdx.x * blockDim.x + threadIdx.x;
  int total = ktiles * ntiles * 512;
  for (; idx < total; idx += gridDim.x * blockDim.x) {
    int i = idx & 15;
    int lane = (idx >> 4) & 31;
    int t = idx >> 9;
    int kk = t / ntiles, nt = t % ntiles;
    int n = lane & 15, hs = lane >> 4;
    dst[idx] = f2bf(src[(size_t)(kk * 32 + hs * 16 + i) * ldn + nt * 16 + n]);
  }
}

// ---------- kernel 2: LayerNorm + shifted window gather -> bf16 [NWIN,64,192] ----------

__global__ void swin_ln_window(const float* __restrict__ xin, const float* __restrict__ g,
                               const float* __restrict__ bta, __bf16* __restrict__ xw, int shift) {
  int lane = threadIdx.x & 31;
  int tw = blockIdx.x * 8 + (threadIdx.x >> 5);   // destination token id: win*64 + n
  int win = tw >> 6, n = tw & 63;
  int b = win >> 8, wy = (win >> 4) & 15, wx = win & 15;
  int iy = n >> 3, ix = n & 7;
  int y  = (wy * 8 + iy + shift) & 127;           // roll(x, -shift) gather
  int xx = (wx * 8 + ix + shift) & 127;
  const float* src = xin + ((((size_t)b * 128 + y) * 128 + xx) * C);

  float v[6], s = 0.f, s2 = 0.f;
#pragma unroll
  for (int j = 0; j < 6; ++j) {
    v[j] = src[lane + 32 * j];
    s += v[j]; s2 += v[j] * v[j];
  }
#pragma unroll
  for (int off = 16; off > 0; off >>= 1) {
    s  += __shfl_xor(s,  off, 32);
    s2 += __shfl_xor(s2, off, 32);
  }
  float mu = s * (1.0f / C);
  float var = s2 * (1.0f / C) - mu * mu;
  float rs = rsqrtf(var + 1e-5f);

  __bf16* dst = xw + (size_t)tw * C;
#pragma unroll
  for (int j = 0; j < 6; ++j) {
    int c = lane + 32 * j;
    dst[c] = f2bf((v[j] - mu) * rs * g[c] + bta[c]);
  }
}

// ---------- kernel 3: fused per-window attention ----------
// One workgroup (256 thr = 8 wave32) per window. Everything stays in LDS.
// LDS map (dynamic): [S f32 6*64*64][qkv bf16 64*576][P bf16 6*64*64][x/o bf16 64*192][rid]

__global__ void swin_window_attn(const __bf16* __restrict__ xw,
                                 const __bf16* __restrict__ wqkv, const float* __restrict__ bqkv,
                                 const __bf16* __restrict__ wproj, const float* __restrict__ bproj,
                                 const float* __restrict__ resid, float* __restrict__ out,
                                 int shift) {
  extern __shared__ char smem[];
  float*  lds_s   = (float*)smem;                                    // 98304 B
  __bf16* lds_qkv = (__bf16*)(smem + 98304);                         // 73728 B
  __bf16* lds_p   = (__bf16*)(smem + 98304 + 73728);                 // 49152 B
  __bf16* lds_x   = (__bf16*)(smem + 98304 + 73728 + 49152);         // 24576 B (reused as O)
  int*    lds_rid = (int*)(smem + 98304 + 73728 + 49152 + 24576);    // 256 B

  const int tid = threadIdx.x;
  const int lane = tid & 31, wave = tid >> 5;
  const int win = blockIdx.x;
  const int bq = win >> 8, wy = (win >> 4) & 15, wx = win & 15;
  const float scale = 0.17677669529663687f;                          // 1/sqrt(32)

  // stage 0: load LN'd window tile (64x192 bf16 = 24 KB) + region ids
  {
    const uint4* src = (const uint4*)(xw + (size_t)win * (NTOK * C));
    uint4* dst = (uint4*)lds_x;
    for (int i = tid; i < (NTOK * C * 2) / 16; i += 256) dst[i] = src[i];
    if (tid < NTOK) {
      int iy = tid >> 3, ix = tid & 7;
      lds_rid[tid] = rcls(wy * 8 + iy) * 3 + rcls(wx * 8 + ix);
    }
  }
  __syncthreads();

  // stage 1: qkv = xw @ Wqkv + b  (M=64, N=576, K=192); q gets 1/sqrt(d) folded in
  for (int job = wave; job < 4 * 36; job += 8) {
    int mt = job / 36, nt = job % 36;
    v8f acc = {};
#pragma unroll
    for (int kk = 0; kk < 6; ++kk) {
      v16bf a = load_A(lds_x, mt * 16, kk * 32, C, lane);
      v16bf b = load_Bsw(wqkv, kk * 36 + nt, lane);
      acc = wmma_bf16(a, b, acc);
    }
    int n = lane & 15, hs = lane >> 4;
    int col = nt * 16 + n;
    float bias = bqkv[col];
    float mul = (col < C) ? scale : 1.0f;
#pragma unroll
    for (int r = 0; r < 8; ++r)
      lds_qkv[(size_t)(mt * 16 + r + 8 * hs) * C3 + col] = f2bf((acc[r] + bias) * mul);
  }
  __syncthreads();

  // stage 2: S_h = q_h @ k_h^T (+ shift mask)  -> lds_s [h][64][64] f32
  for (int job = wave; job < HEADS * 16; job += 8) {
    int h = job / 16, t = job % 16, mt = t / 4, nt = t % 4;
    v16bf a = load_A(lds_qkv, mt * 16, h * 32, C3, lane);
    v16bf b = load_Bt(lds_qkv + C + h * 32, nt * 16, C3, lane);
    v8f acc = {};
    acc = wmma_bf16(a, b, acc);
    int n = lane & 15, hs = lane >> 4;
    int col = nt * 16 + n;
#pragma unroll
    for (int r = 0; r < 8; ++r) {
      int row = mt * 16 + r + 8 * hs;
      float v = acc[r];
      if (shift != 0 && lds_rid[row] != lds_rid[col]) v -= 100.0f;
      lds_s[(size_t)h * 4096 + row * 64 + col] = v;
    }
  }
  __syncthreads();

  // stage 3: row softmax over 384 rows of 64 -> lds_p bf16
  for (int row = tid; row < HEADS * NTOK; row += 256) {
    float* s = lds_s + (size_t)row * 64;
    float mx = -3.4e38f;
#pragma unroll 4
    for (int m = 0; m < 64; ++m) mx = fmaxf(mx, s[m]);
    float sum = 0.f;
#pragma unroll 4
    for (int m = 0; m < 64; ++m) { float e = __expf(s[m] - mx); s[m] = e; sum += e; }
    float inv = 1.0f / sum;
    __bf16* p = lds_p + (size_t)row * 64;
#pragma unroll 4
    for (int m = 0; m < 64; ++m) p[m] = f2bf(s[m] * inv);
  }
  __syncthreads();

  // stage 4: O_h = P_h @ v_h  (M=64, N=32, K=64) -> lds_o (= lds_x) [64][192]
  __bf16* lds_o = lds_x;
  for (int job = wave; job < HEADS * 8; job += 8) {
    int h = job / 8, t = job % 8, mt = t / 2, nt = t % 2;
    v8f acc = {};
#pragma unroll
    for (int kk = 0; kk < 2; ++kk) {
      v16bf a = load_A(lds_p + (size_t)h * 4096, mt * 16, kk * 32, 64, lane);
      v16bf b = load_B_lds(lds_qkv + 2 * C + h * 32, kk * 32, nt * 16, C3, lane);
      acc = wmma_bf16(a, b, acc);
    }
    store_C_bf(lds_o, mt * 16, h * 32 + nt * 16, C, lane, acc);
  }
  __syncthreads();

  // stage 5: out = O @ Wproj + b + residual, scatter with inverse shift
  for (int job = wave; job < 4 * 12; job += 8) {
    int mt = job / 12, nt = job % 12;
    v8f acc = {};
#pragma unroll
    for (int kk = 0; kk < 6; ++kk) {
      v16bf a = load_A(lds_o, mt * 16, kk * 32, C, lane);
      v16bf b = load_Bsw(wproj, kk * 12 + nt, lane);
      acc = wmma_bf16(a, b, acc);
    }
    int n = lane & 15, hs = lane >> 4;
    int col = nt * 16 + n;
    float bias = bproj[col];
#pragma unroll
    for (int r = 0; r < 8; ++r) {
      int row = mt * 16 + r + 8 * hs;               // token in window
      int iy = row >> 3, ix = row & 7;
      int y  = (wy * 8 + iy + shift) & 127;         // roll(out, +shift) scatter
      int xx = (wx * 8 + ix + shift) & 127;
      size_t idx = ((((size_t)bq * 128 + y) * 128 + xx) * C) + col;
      out[idx] = resid[idx] + acc[r] + bias;
    }
  }
}

// ---------- launch ----------

extern "C" void kernel_launch(void* const* d_in, const int* in_sizes, int n_in,
                              void* d_out, int out_size, void* d_ws, size_t ws_size,
                              hipStream_t stream) {
  const float* x      = (const float*)d_in[0];
  const float* ln1_g  = (const float*)d_in[1];
  const float* ln1_b  = (const float*)d_in[2];
  const float* qkv1_w = (const float*)d_in[3];
  const float* qkv1_b = (const float*)d_in[4];
  const float* prj1_w = (const float*)d_in[5];
  const float* prj1_b = (const float*)d_in[6];
  const float* ln2_g  = (const float*)d_in[7];
  const float* ln2_b  = (const float*)d_in[8];
  const float* qkv2_w = (const float*)d_in[9];
  const float* qkv2_b = (const float*)d_in[10];
  const float* prj2_w = (const float*)d_in[11];
  const float* prj2_b = (const float*)d_in[12];
  float* out = (float*)d_out;

  char* ws = (char*)d_ws;
  __bf16* wqkv1 = (__bf16*)(ws);
  __bf16* wprj1 = (__bf16*)(ws + 221184);
  __bf16* wqkv2 = (__bf16*)(ws + 221184 + 73728);
  __bf16* wprj2 = (__bf16*)(ws + 2 * 221184 + 73728);
  __bf16* xw    = (__bf16*)(ws + 2 * 221184 + 2 * 73728);   // 2048*64*192 bf16 = 50.3 MB

  const int NQKV = C * C3, NPRJ = C * C;
  swin_swizzle_w<<<(NQKV + 255) / 256, 256, 0, stream>>>(qkv1_w, wqkv1, 6, 36, C3);
  swin_swizzle_w<<<(NPRJ + 255) / 256, 256, 0, stream>>>(prj1_w, wprj1, 6, 12, C);
  swin_swizzle_w<<<(NQKV + 255) / 256, 256, 0, stream>>>(qkv2_w, wqkv2, 6, 36, C3);
  swin_swizzle_w<<<(NPRJ + 255) / 256, 256, 0, stream>>>(prj2_w, wprj2, 6, 12, C);

  size_t smem = 98304 + 73728 + 49152 + 24576 + 256;  // 246016 B (< 320 KB WGP LDS)

  // branch 1: W-MSA (shift = 0)
  swin_ln_window<<<NWIN * 64 / 8, 256, 0, stream>>>(x, ln1_g, ln1_b, xw, 0);
  swin_window_attn<<<NWIN, 256, smem, stream>>>(xw, wqkv1, qkv1_b, wprj1, prj1_b, x, out, 0);

  // branch 2: SW-MSA (shift = 4), in-place on d_out (bijective token ownership)
  swin_ln_window<<<NWIN * 64 / 8, 256, 0, stream>>>(out, ln2_g, ln2_b, xw, 4);
  swin_window_attn<<<NWIN, 256, smem, stream>>>(xw, wqkv2, qkv2_b, wprj2, prj2_b, out, out, 4);
}
